// QuantTrinaryFCMNIST_7971459301379
// MI455X (gfx1250) — 8-bit matrix operand order repaired and checked by reading, compiled, not run on hardware
//
#include <hip/hip_runtime.h>

typedef signed char s8;
typedef unsigned long long u64;
typedef __attribute__((ext_vector_type(8))) int v8i;

#define BATCH 16384
#define K1    784
#define K1P   832
#define N1    1024
#define K2    1024
#define N2P   16
#define NOUT  10

#define OFF_SUMS1   0u
#define OFF_SUMS2   16384u
#define OFF_TSTAT   16640u
#define STATS_U64   2112u
#define OFF_QX      16896u
#define OFF_QW1     (OFF_QX   + 16384u*832u)
#define OFF_QW2     (OFF_QW1  + 1024u*832u)
#define OFF_C1      (OFF_QW2  + 16u*1024u)
#define OFF_QH      (OFF_C1   + 16384u*1024u*2u)
#define OFF_C2      (OFF_QH   + 16384u*1024u)
#define OFF_TBUF    (OFF_C2   + 16384u*16u*4u)
#define OFF_BN1S    (OFF_TBUF + 16384u*16u)
#define OFF_BN1B    (OFF_BN1S + 4096u)
#define OFF_BN2S    (OFF_BN1B + 4096u)
#define OFF_BN2B    (OFF_BN2S + 64u)

__device__ __forceinline__ int clamp1(int q) { return q < -1 ? -1 : (q > 1 ? 1 : q); }

__device__ __forceinline__ v8i load_a_frag(const int* __restrict__ rowPtr, int kb, int half) {
    v8i a;
    #pragma unroll
    for (int i = 0; i < 8; ++i) {
        int k = kb + ((i >> 1) << 4) + ((i & 1) << 2) + (half << 3);
        a[i] = rowPtr[k >> 2];
    }
    return a;
}

__device__ __forceinline__ v8i load_b_frag(const int* __restrict__ rowPtr, int kb, int half) {
    v8i b;
    #pragma unroll
    for (int v = 0; v < 8; ++v) {
        int k = kb + ((v >> 1) << 4) + ((v & 1) << 2) + (half << 3);
        b[v] = rowPtr[k >> 2];
    }
    return b;
}

__global__ void zero_u64_kernel(u64* p, int n) {
    int i = blockIdx.x * blockDim.x + threadIdx.x;
    if (i < n) p[i] = 0ull;
}

__global__ __launch_bounds__(256) void quant_x_kernel(const float* __restrict__ x, int* __restrict__ qx) {
    unsigned idx = blockIdx.x * 256u + threadIdx.x;
    unsigned r = idx / 208u, d = idx % 208u;
    int out = 0;
    #pragma unroll
    for (int j = 0; j < 4; ++j) {
        int k = (int)(d * 4u) + j;
        int q = 0;
        if (k < K1) {
            float v = x[(size_t)r * K1 + k];
            q = clamp1((int)rintf(2.0f * v - 1.0f));
        }
        out |= (q & 0xFF) << (8 * j);
    }
    qx[idx] = out;
}

__global__ __launch_bounds__(256) void quant_w1_kernel(const float* __restrict__ w, int* __restrict__ qw) {
    unsigned idx = blockIdx.x * 256u + threadIdx.x;
    unsigned r = idx / 208u, d = idx % 208u;
    int out = 0;
    #pragma unroll
    for (int j = 0; j < 4; ++j) {
        int k = (int)(d * 4u) + j;
        int q = 0;
        if (k < K1) q = clamp1((int)rintf(w[(size_t)r * K1 + k]));
        out |= (q & 0xFF) << (8 * j);
    }
    qw[idx] = out;
}

__global__ __launch_bounds__(256) void quant_w2_kernel(const float* __restrict__ w, int* __restrict__ qw) {
    unsigned idx = blockIdx.x * 256u + threadIdx.x;
    unsigned r = idx / 256u, d = idx % 256u;
    int out = 0;
    if (r < NOUT) {
        #pragma unroll
        for (int j = 0; j < 4; ++j) {
            int k = (int)(d * 4u) + j;
            int q = clamp1((int)rintf(w[(size_t)r * K2 + k]));
            out |= (q & 0xFF) << (8 * j);
        }
    }
    qw[idx] = out;
}

__global__ __launch_bounds__(256) void gemm1_kernel(const s8* __restrict__ qx,
                                                    const s8* __restrict__ qw,
                                                    short* __restrict__ c1) {
    const int tid   = threadIdx.x;
    const int lane  = tid & 31;
    const int wave  = tid >> 5;
    const int id    = blockIdx.x * 8 + wave;
    const int mTile = id >> 4;
    const int nGrp  = id & 15;
    const int half  = lane >> 4;
    const int l15   = lane & 15;
    const int mBase = mTile * 32;
    const int nBase = nGrp * 64;

    const int* aP0 = (const int*)(qx + (size_t)(mBase + l15) * K1P);
    const int* aP1 = (const int*)(qx + (size_t)(mBase + 16 + l15) * K1P);
    const int* bP0 = (const int*)(qw + (size_t)(nBase +  0 + l15) * K1P);
    const int* bP1 = (const int*)(qw + (size_t)(nBase + 16 + l15) * K1P);
    const int* bP2 = (const int*)(qw + (size_t)(nBase + 32 + l15) * K1P);
    const int* bP3 = (const int*)(qw + (size_t)(nBase + 48 + l15) * K1P);

    v8i acc[8];
    #pragma unroll
    for (int i = 0; i < 8; ++i) acc[i] = (v8i){0,0,0,0,0,0,0,0};

    v8i a0 = load_a_frag(aP0, 0, half);
    v8i a1 = load_a_frag(aP1, 0, half);
    v8i b0 = load_b_frag(bP0, 0, half);
    v8i b1 = load_b_frag(bP1, 0, half);
    v8i b2 = load_b_frag(bP2, 0, half);
    v8i b3 = load_b_frag(bP3, 0, half);

    #pragma unroll 2
    for (int kb = 0; kb < K1P - 64; kb += 64) {
        const int kn = kb + 64;
        v8i na0 = load_a_frag(aP0, kn, half);
        v8i na1 = load_a_frag(aP1, kn, half);
        v8i nb0 = load_b_frag(bP0, kn, half);
        v8i nb1 = load_b_frag(bP1, kn, half);
        v8i nb2 = load_b_frag(bP2, kn, half);
        v8i nb3 = load_b_frag(bP3, kn, half);

        acc[0] = __builtin_amdgcn_wmma_i32_16x16x64_iu8(true, a0, true, b0, acc[0], false, false);
        acc[1] = __builtin_amdgcn_wmma_i32_16x16x64_iu8(true, a0, true, b1, acc[1], false, false);
        acc[2] = __builtin_amdgcn_wmma_i32_16x16x64_iu8(true, a0, true, b2, acc[2], false, false);
        acc[3] = __builtin_amdgcn_wmma_i32_16x16x64_iu8(true, a0, true, b3, acc[3], false, false);
        acc[4] = __builtin_amdgcn_wmma_i32_16x16x64_iu8(true, a1, true, b0, acc[4], false, false);
        acc[5] = __builtin_amdgcn_wmma_i32_16x16x64_iu8(true, a1, true, b1, acc[5], false, false);
        acc[6] = __builtin_amdgcn_wmma_i32_16x16x64_iu8(true, a1, true, b2, acc[6], false, false);
        acc[7] = __builtin_amdgcn_wmma_i32_16x16x64_iu8(true, a1, true, b3, acc[7], false, false);

        a0 = na0; a1 = na1; b0 = nb0; b1 = nb1; b2 = nb2; b3 = nb3;
    }

    acc[0] = __builtin_amdgcn_wmma_i32_16x16x64_iu8(true, a0, true, b0, acc[0], false, false);
    acc[1] = __builtin_amdgcn_wmma_i32_16x16x64_iu8(true, a0, true, b1, acc[1], false, false);
    acc[2] = __builtin_amdgcn_wmma_i32_16x16x64_iu8(true, a0, true, b2, acc[2], false, false);
    acc[3] = __builtin_amdgcn_wmma_i32_16x16x64_iu8(true, a0, true, b3, acc[3], false, false);
    acc[4] = __builtin_amdgcn_wmma_i32_16x16x64_iu8(true, a1, true, b0, acc[4], false, false);
    acc[5] = __builtin_amdgcn_wmma_i32_16x16x64_iu8(true, a1, true, b1, acc[5], false, false);
    acc[6] = __builtin_amdgcn_wmma_i32_16x16x64_iu8(true, a1, true, b2, acc[6], false, false);
    acc[7] = __builtin_amdgcn_wmma_i32_16x16x64_iu8(true, a1, true, b3, acc[7], false, false);

    #pragma unroll
    for (int mt = 0; mt < 2; ++mt) {
        #pragma unroll
        for (int nt = 0; nt < 4; ++nt) {
            #pragma unroll
            for (int v = 0; v < 8; ++v) {
                int row = mBase + mt * 16 + v + (half << 3);
                int col = nBase + nt * 16 + l15;
                c1[(size_t)row * N1 + col] = (short)acc[mt * 4 + nt][v];
            }
        }
    }
}

__global__ __launch_bounds__(256) void stats1_kernel(const short* __restrict__ c1,
                                                     u64* __restrict__ sums) {
    int col = blockIdx.x * 256 + threadIdx.x;
    int r0  = blockIdx.y * 512;
    int s = 0;
    long long sq = 0;
    for (int i = 0; i < 512; ++i) {
        int v = c1[(size_t)(r0 + i) * N1 + col];
        s += v;
        sq += v * v;
    }
    atomicAdd(&sums[col * 2 + 0], (u64)(long long)s);
    atomicAdd(&sums[col * 2 + 1], (u64)sq);
}

__global__ void bn_prep_kernel(const u64* __restrict__ sums,
                               const float* __restrict__ gamma,
                               const float* __restrict__ beta,
                               float* __restrict__ scale, float* __restrict__ shift,
                               int ncols, int nvalid, float eps, double invN) {
    int c = blockIdx.x * blockDim.x + threadIdx.x;
    if (c >= ncols) return;
    if (c >= nvalid) { scale[c] = 0.0f; shift[c] = 0.0f; return; }
    double s    = (double)(long long)sums[c * 2 + 0];
    double sq   = (double)(long long)sums[c * 2 + 1];
    double mean = s * invN;
    double var  = sq * invN - mean * mean;
    double rstd = 1.0 / sqrt(var + (double)eps);
    double g    = (double)gamma[c];
    scale[c] = (float)(g * rstd);
    shift[c] = (float)((double)beta[c] - mean * g * rstd);
}

__global__ __launch_bounds__(256) void bn1_apply_kernel(const short* __restrict__ c1,
                                                        const float* __restrict__ scale,
                                                        const float* __restrict__ shift,
                                                        int* __restrict__ qh) {
    unsigned idx = blockIdx.x * 256u + threadIdx.x;
    unsigned r = idx >> 8, d = idx & 255u;
    const short* row = c1 + (size_t)r * N1 + d * 4u;
    int out = 0;
    #pragma unroll
    for (int j = 0; j < 4; ++j) {
        int col = (int)(d * 4u) + j;
        float y = (float)row[j] * scale[col] + shift[col];
        int q = clamp1((int)rintf(y));
        out |= (q & 0xFF) << (8 * j);
    }
    qh[idx] = out;
}

__global__ __launch_bounds__(256) void gemm2_kernel(const s8* __restrict__ qh,
                                                    const s8* __restrict__ qw2,
                                                    int* __restrict__ c2) {
    const int tid   = threadIdx.x;
    const int lane  = tid & 31;
    const int wave  = tid >> 5;
    const int id    = blockIdx.x * 8 + wave;
    const int half  = lane >> 4;
    const int l15   = lane & 15;
    const int mBase = id * 32;

    const int* aP0 = (const int*)(qh  + (size_t)(mBase + l15) * K2);
    const int* aP1 = (const int*)(qh  + (size_t)(mBase + 16 + l15) * K2);
    const int* bP  = (const int*)(qw2 + (size_t)l15 * K2);

    v8i acc0 = (v8i){0,0,0,0,0,0,0,0};
    v8i acc1 = (v8i){0,0,0,0,0,0,0,0};

    v8i a0 = load_a_frag(aP0, 0, half);
    v8i a1 = load_a_frag(aP1, 0, half);
    v8i b  = load_b_frag(bP,  0, half);

    #pragma unroll 2
    for (int kb = 0; kb < K2 - 64; kb += 64) {
        const int kn = kb + 64;
        v8i na0 = load_a_frag(aP0, kn, half);
        v8i na1 = load_a_frag(aP1, kn, half);
        v8i nb  = load_b_frag(bP,  kn, half);
        acc0 = __builtin_amdgcn_wmma_i32_16x16x64_iu8(true, a0, true, b, acc0, false, false);
        acc1 = __builtin_amdgcn_wmma_i32_16x16x64_iu8(true, a1, true, b, acc1, false, false);
        a0 = na0; a1 = na1; b = nb;
    }
    acc0 = __builtin_amdgcn_wmma_i32_16x16x64_iu8(true, a0, true, b, acc0, false, false);
    acc1 = __builtin_amdgcn_wmma_i32_16x16x64_iu8(true, a1, true, b, acc1, false, false);

    #pragma unroll
    for (int v = 0; v < 8; ++v) {
        c2[(size_t)(mBase + v + (half << 3)) * N2P + l15]      = acc0[v];
        c2[(size_t)(mBase + 16 + v + (half << 3)) * N2P + l15] = acc1[v];
    }
}

__global__ __launch_bounds__(256) void stats2_kernel(const int* __restrict__ c2,
                                                     u64* __restrict__ sums) {
    int tid  = threadIdx.x;
    int col  = tid & 15;
    int rb   = blockIdx.x * 256 + (tid >> 4);
    long long s = 0, sq = 0;
    for (int i = 0; i < 16; ++i) {
        int v = c2[(size_t)(rb + i * 16) * N2P + col];
        s += v;
        sq += (long long)v * v;
    }
    atomicAdd(&sums[col * 2 + 0], (u64)s);
    atomicAdd(&sums[col * 2 + 1], (u64)sq);
}

__global__ __launch_bounds__(256) void final_a_kernel(const int* __restrict__ c2,
                                                      const float* __restrict__ scale,
                                                      const float* __restrict__ shift,
                                                      s8* __restrict__ tbuf,
                                                      int* __restrict__ tstats) {
    __shared__ int ss[2];
    if (threadIdx.x == 0) { ss[0] = 0; ss[1] = 0; }
    __syncthreads();
    unsigned idx = blockIdx.x * 256u + threadIdx.x;
    unsigned r = idx / (unsigned)NOUT, col = idx % (unsigned)NOUT;
    int c = c2[(size_t)r * N2P + col];
    float y = (float)c * scale[col] + shift[col];
    int t = clamp1((int)rintf(y));
    tbuf[(size_t)r * N2P + col] = (s8)t;
    atomicAdd(&ss[0], t);
    atomicAdd(&ss[1], t * t);
    __syncthreads();
    if (threadIdx.x == 0) {
        atomicAdd(&tstats[0], ss[0]);
        atomicAdd(&tstats[1], ss[1]);
    }
}

__global__ __launch_bounds__(256) void final_b_kernel(const s8* __restrict__ tbuf,
                                                      const int* __restrict__ tstats,
                                                      const float* __restrict__ tw,
                                                      const float* __restrict__ tb,
                                                      float* __restrict__ out) {
    unsigned idx = blockIdx.x * 256u + threadIdx.x;
    unsigned r = idx / (unsigned)NOUT, col = idx % (unsigned)NOUT;
    const double N = (double)(BATCH * NOUT);
    double mean = (double)tstats[0] / N;
    double var  = ((double)tstats[1] - N * mean * mean) / (N - 1.0);
    float inv   = (float)(1.0 / sqrt(var + 1e-4));
    float t     = (float)tbuf[(size_t)r * N2P + col];
    out[idx] = (t - (float)mean) * inv * tw[0] + tb[0];
}

extern "C" void kernel_launch(void* const* d_in, const int* in_sizes, int n_in,
                              void* d_out, int out_size, void* d_ws, size_t ws_size,
                              hipStream_t stream) {
    (void)in_sizes; (void)n_in; (void)out_size; (void)ws_size;
    const float* x      = (const float*)d_in[0];
    const float* W1     = (const float*)d_in[1];
    const float* gamma1 = (const float*)d_in[2];
    const float* beta1  = (const float*)d_in[3];
    const float* W2     = (const float*)d_in[4];
    const float* gamma2 = (const float*)d_in[5];
    const float* beta2  = (const float*)d_in[6];
    const float* tn_w   = (const float*)d_in[7];
    const float* tn_b   = (const float*)d_in[8];

    char* ws = (char*)d_ws;
    u64*   sums1  = (u64*)  (ws + OFF_SUMS1);
    u64*   sums2  = (u64*)  (ws + OFF_SUMS2);
    int*   tstats = (int*)  (ws + OFF_TSTAT);
    s8*    qx     = (s8*)   (ws + OFF_QX);
    s8*    qw1    = (s8*)   (ws + OFF_QW1);
    s8*    qw2    = (s8*)   (ws + OFF_QW2);
    short* c1     = (short*)(ws + OFF_C1);
    s8*    qh     = (s8*)   (ws + OFF_QH);
    int*   c2     = (int*)  (ws + OFF_C2);
    s8*    tbuf   = (s8*)   (ws + OFF_TBUF);
    float* bn1s   = (float*)(ws + OFF_BN1S);
    float* bn1b   = (float*)(ws + OFF_BN1B);
    float* bn2s   = (float*)(ws + OFF_BN2S);
    float* bn2b   = (float*)(ws + OFF_BN2B);

    zero_u64_kernel<<<9, 256, 0, stream>>>((u64*)ws, (int)STATS_U64);

    quant_x_kernel <<<(BATCH * 208) / 256, 256, 0, stream>>>(x,  (int*)qx);
    quant_w1_kernel<<<(N1   * 208) / 256, 256, 0, stream>>>(W1, (int*)qw1);
    quant_w2_kernel<<<16,                256, 0, stream>>>(W2, (int*)qw2);

    gemm1_kernel<<<1024, 256, 0, stream>>>(qx, qw1, c1);

    stats1_kernel<<<dim3(4, 32), 256, 0, stream>>>(c1, sums1);
    bn_prep_kernel<<<4, 256, 0, stream>>>(sums1, gamma1, beta1, bn1s, bn1b,
                                          N1, N1, 1e-5f, 1.0 / (double)BATCH);
    bn1_apply_kernel<<<BATCH, 256, 0, stream>>>(c1, bn1s, bn1b, (int*)qh);

    gemm2_kernel<<<64, 256, 0, stream>>>(qh, qw2, c2);

    stats2_kernel<<<64, 256, 0, stream>>>(c2, sums2);
    bn_prep_kernel<<<1, 32, 0, stream>>>(sums2, gamma2, beta2, bn2s, bn2b,
                                         N2P, NOUT, 1e-5f, 1.0 / (double)BATCH);

    final_a_kernel<<<640, 256, 0, stream>>>(c2, bn2s, bn2b, tbuf, tstats);
    final_b_kernel<<<640, 256, 0, stream>>>(tbuf, tstats, tn_w, tn_b, (float*)d_out);
}
